// GraphPropagationAttention_3384434229997
// MI455X (gfx1250) — matrix operand order repaired and checked by reading, not compiled, not run on hardware
//
#include <hip/hip_runtime.h>
#include <hip/hip_bf16.h>

#define BB 4
#define CC 256
#define NN 1024
#define HH 8
#define DD 32
#define EPS 1e-5f
#define QSCALE 0.17677669529663687f   // 1/sqrt(D)

typedef __attribute__((ext_vector_type(16))) _Float16 v16h;
typedef __attribute__((ext_vector_type(8)))  _Float16 half8;
typedef __attribute__((ext_vector_type(8)))  float    v8f;
typedef unsigned int v4u __attribute__((ext_vector_type(4)));
typedef int          v8i __attribute__((ext_vector_type(8)));
typedef int          v4i __attribute__((ext_vector_type(4)));

#if defined(__has_builtin)
#if __has_builtin(__builtin_amdgcn_tensor_load_to_lds) && \
    __has_builtin(__builtin_amdgcn_s_wait_tensorcnt)
#define USE_TDM 1
#endif
#if __has_builtin(__builtin_amdgcn_sched_group_barrier)
// masks: 0x8 = MFMA/WMMA, 0x20 = VMEM read
#define SGB(mask, n) __builtin_amdgcn_sched_group_barrier(mask, n, 0)
#define HAVE_SGB 1
#endif
#endif
#ifndef SGB
#define SGB(mask, n)
#endif

// Pin pipeline: first WMMA waits only after 2 fragments are in flight.
// npre = loads in first group; then (2-load, 1-wmma) x 6, then trailing wmma.
#define PIPE8(npre)                                                     \
  SGB(0x20, npre); SGB(0x8, 1);                                         \
  SGB(0x20, 2); SGB(0x8, 1); SGB(0x20, 2); SGB(0x8, 1);                 \
  SGB(0x20, 2); SGB(0x8, 1); SGB(0x20, 2); SGB(0x8, 1);                 \
  SGB(0x20, 2); SGB(0x8, 1); SGB(0x20, 2); SGB(0x8, 1);                 \
  SGB(0x8, 1)

__device__ __forceinline__ v8f wmma16x16x32(v16h a, v16h b, v8f c) {
  // D = A(16x32 f16) x B(32x16 f16) + C(16x16 f32)
  return __builtin_amdgcn_wmma_f32_16x16x32_f16(false, a, false, b, (short)0, c,
                                                false, false);
}

// B-fragment: lane (l&15) selects column; contiguous 16 halves along K,
// lanes >=16 take K+16 (caller applies the +16 offset in the pointer).
__device__ __forceinline__ v16h load_bfrag(const _Float16* p) {
  v16h r;
  half8* rp = reinterpret_cast<half8*>(&r);
  rp[0] = *reinterpret_cast<const half8*>(p);
  rp[1] = *reinterpret_cast<const half8*>(p + 8);
  return r;
}

// A-fragment (row-major 16x32): lane<16 -> K {0..15}, lane>=16 -> K {16..31}
__device__ __forceinline__ v16h load_afrag(const _Float16* rowptr, int lane) {
  const int ko = (lane & 16) ? 16 : 0;
  v16h r;
  half8* rp = reinterpret_cast<half8*>(&r);
  rp[0] = *reinterpret_cast<const half8*>(rowptr + ko);
  rp[1] = *reinterpret_cast<const half8*>(rowptr + 8 + ko);
  return r;
}

// ---------------------------------------------------------------- prep kernels
__global__ void k_prior(const float* __restrict__ da, float* __restrict__ prior) {
  const int b = blockIdx.x, c = threadIdx.x;
  float s = 0.f;
  const float* p = da + (size_t)b * NN * CC + c;
  for (int n = 0; n < NN; ++n) s += p[(size_t)n * CC];
  prior[b * CC + c] = s;                      // sum over N of da_prior[b,:,c]
}

__global__ void k_wq(const float* __restrict__ w, _Float16* __restrict__ wT) {
  const int j = blockIdx.x, c = threadIdx.x;  // wT[j][c] = w_qkv[c][j]
  wT[(size_t)j * CC + c] = (_Float16)w[(size_t)c * (3 * CC) + j];
}

__global__ void k_wp(const float* __restrict__ w, _Float16* __restrict__ wT) {
  const int j = blockIdx.x, c = threadIdx.x;  // wT[j][c] = w_proj[c][j]
  wT[(size_t)j * CC + c] = (_Float16)w[(size_t)c * CC + j];
}

// --------------------------------------------- diag*prior residual + LayerNorm
__global__ void k_ln(const float* __restrict__ ne, const float* __restrict__ edge,
                     const float* __restrict__ prior, const float* __restrict__ gamma,
                     const float* __restrict__ beta, _Float16* __restrict__ lnb) {
  const int row = blockIdx.x;                 // b*N + n
  const int b = row >> 10, n = row & (NN - 1);
  const int c = threadIdx.x;
  const float diag = edge[((size_t)(b * NN + n)) * NN + n];
  const float x = ne[((size_t)(b * CC + c)) * NN + n];
  const float x2 = diag * x * prior[b * CC + c] + x;
  __shared__ float rs[CC], rq[CC];
  rs[c] = x2; rq[c] = x2 * x2; __syncthreads();
  for (int s = CC / 2; s > 0; s >>= 1) {
    if (c < s) { rs[c] += rs[c + s]; rq[c] += rq[c + s]; }
    __syncthreads();
  }
  const float mu = rs[0] * (1.f / CC);
  const float var = rq[0] * (1.f / CC) - mu * mu;
  const float v = (x2 - mu) * rsqrtf(var + EPS) * gamma[c] + beta[c];
  lnb[(size_t)row * CC + c] = (_Float16)v;
}

// ------------------------------------------------------------- QKV GEMM (WMMA)
__global__ void __launch_bounds__(256) k_qkv(const _Float16* __restrict__ lnb,
                                             const _Float16* __restrict__ wT,
                                             _Float16* __restrict__ qb,
                                             _Float16* __restrict__ kb,
                                             _Float16* __restrict__ vT) {
  const int rb = blockIdx.x;
  const int b = rb >> 6, n0 = (rb & 63) << 4;
  const int tid = threadIdx.x, lane = tid & 31;
  const int wv = __builtin_amdgcn_readfirstlane(tid >> 5);   // SGPR wave id
  const _Float16* arow = lnb + ((size_t)(b * NN + n0 + (lane & 15))) * CC;
  for (int t = 0; t < 6; ++t) {               // 8 waves x 6 tiles = 48 = 3C/16
    const int jt = wv * 6 + t;
    const _Float16* bbase = wT + (size_t)(jt * 16 + (lane & 15)) * CC +
                            ((lane & 16) ? 16 : 0);
    v8f acc = {};
    v16h bfA = load_bfrag(bbase);
    v16h bfB = bfA;
#pragma unroll
    for (int kc = 0; kc < 8; ++kc) {          // C/32 K-steps
      v16h a = load_afrag(arow + kc * 32, lane);
      if (kc & 1) {
        if (kc < 7) bfA = load_bfrag(bbase + (kc + 1) * 32);
        acc = wmma16x16x32(a, bfB, acc);
      } else {
        if (kc < 7) bfB = load_bfrag(bbase + (kc + 1) * 32);
        acc = wmma16x16x32(a, bfA, acc);
      }
    }
    PIPE8(4);   // 16 b128 loads interleaved one fragment ahead of 8 WMMAs
    // wave-uniform routing: j = jt*16+col -> sel/h/d0 depend only on jt (SGPR)
    const int sel = jt >> 4;
    const int h   = (jt >> 1) & 7;
    const int d0  = (jt & 1) << 4;
    const int col = lane & 15;
    const int d   = d0 + col;
#pragma unroll
    for (int r = 0; r < 8; ++r) {
      const int n = n0 + r + ((lane & 16) ? 8 : 0);
      const float v = acc[r];
      if (sel == 0)
        qb[(((size_t)(b * HH + h)) * NN + n) * DD + d] = (_Float16)(v * QSCALE);
      else if (sel == 1)
        kb[(((size_t)(b * HH + h)) * NN + n) * DD + d] = (_Float16)v;
      else
        vT[(((size_t)(b * HH + h)) * DD + d) * NN + n] = (_Float16)v;  // transposed
    }
  }
}

// --------------------- fused attention: scores+bias, softmax, A@V, edge reduce
__global__ void __launch_bounds__(256, 1)
k_attn(const _Float16* __restrict__ qb, const _Float16* __restrict__ kb,
       const _Float16* __restrict__ vT, const float* __restrict__ edge_in,
       const float* __restrict__ w_expand, const float* __restrict__ b_expand,
       const float* __restrict__ w_reduce, const float* __restrict__ b_reduce,
       float* __restrict__ node_acc, float* __restrict__ wsum_ws,
       float* __restrict__ edge_out) {
  extern __shared__ float smem[];
  float*    S      = smem;                 // 16x1024 f32 scores / exp
  float*    E      = S + 16 * NN;          // 16x1024 f32 edge accumulator
  float*    ET     = E + 16 * NN;          // 16x1024 f32 edge-bias tile (LDS byte off 131072)
  float*    O      = ET + 16 * NN;         // 16x32 per-head output tile
  float*    red    = O + 512;              // 256-wide reduce scratch
  float*    rowred = red + 256;            // 16 per-row results
  _Float16* Sh     = (_Float16*)(rowred + 16);  // 16x1024 f16 probs (A-frag source)

  const int rb = blockIdx.x;
  const int b = rb >> 6, n0 = (rb & 63) << 4;
  const int tid = threadIdx.x, lane = tid & 31;
  const int wv = __builtin_amdgcn_readfirstlane(tid >> 5);   // SGPR wave id

#ifdef USE_TDM
  // Tensor Data Mover: DMA the 16x1024 f32 edge tile (row stride N) into ET.
  if (wv == 0) {
    const unsigned long long ga =
        (unsigned long long)(const void*)(edge_in + (size_t)(b * NN + n0) * NN);
    const unsigned lds_addr = 2u * 16u * NN * 4u;   // ET offset in dynamic LDS (base 0)
    v4u g0;
    g0.x = 1u;                                       // count=1 (valid user D#)
    g0.y = lds_addr;                                 // D#.lds_addr (bytes)
    g0.z = (unsigned)(ga & 0xFFFFFFFFull);           // D#.global_addr[31:0]
    g0.w = (unsigned)((ga >> 32) & 0x01FFFFFFull)    // D#.global_addr[56:32]
           | (2u << 30);                             // D#.type=2 ("image")
    v8i g1;
    g1[0] = 0x00020000;                              // data_size=2 (4 bytes)
    g1[1] = (int)((NN & 0xFFFF) << 16);              // tensor_dim0 = 1024 (lo16)
    g1[2] = (int)(16u << 16);                        // tensor_dim0 hi=0 | tensor_dim1=16
    g1[3] = (int)((unsigned)NN << 16);               // tensor_dim1 hi=0 | tile_dim0=1024
    g1[4] = 16;                                      // tile_dim1=16, tile_dim2=0
    g1[5] = NN;                                      // tensor_dim0_stride lo32 = 1024
    g1[6] = 0;                                       // stride hi | dim1_stride lo
    g1[7] = 0;
    v4i gz = {0, 0, 0, 0};
#if __clang_major__ >= 23
    v8i gz8 = {};
    __builtin_amdgcn_tensor_load_to_lds(g0, g1, gz, gz, gz8, 0);
#else
    __builtin_amdgcn_tensor_load_to_lds(g0, g1, gz, gz, 0);
#endif
    __builtin_amdgcn_s_wait_tensorcnt(0);
  }
  for (int idx = tid; idx < 16 * NN; idx += 256) E[idx] = 0.f;
#else
  for (int idx = tid; idx < 16 * NN; idx += 256) {
    ET[idx] = edge_in[((size_t)(b * NN + n0 + (idx >> 10))) * NN + (idx & (NN - 1))];
    E[idx] = 0.f;
  }
#endif
  __syncthreads();

  const int srow = tid >> 4, ssub = tid & 15, sbase = ssub * 64;

  for (int h = 0; h < HH; ++h) {
    const float we = w_expand[h], be = b_expand[h], wr = w_reduce[h];
    O[tid] = 0.f; O[tid + 256] = 0.f;
    // Q A-fragment for this head's 16 rows
    const _Float16* qrow =
        qb + (((size_t)(b * HH + h)) * NN + n0 + (lane & 15)) * DD;
    v16h aq = load_afrag(qrow, lane);
    // scores: wave wv owns 8 column tiles of 16; pipelined K fragments
    const _Float16* kbase = kb + (((size_t)(b * HH + h)) * NN + (lane & 15)) * DD +
                            ((lane & 16) ? 16 : 0);
    v16h kfA = load_bfrag(kbase + (size_t)(wv * 8) * 16 * DD);
    v16h kfB = kfA;
#pragma unroll
    for (int t = 0; t < 8; ++t) {
      const int m0 = (wv * 8 + t) * 16;
      v8f c = {};
      if (t & 1) {
        if (t < 7) kfA = load_bfrag(kbase + (size_t)(m0 + 16) * DD);
        c = wmma16x16x32(aq, kfB, c);
      } else {
        if (t < 7) kfB = load_bfrag(kbase + (size_t)(m0 + 16) * DD);
        c = wmma16x16x32(aq, kfA, c);
      }
      if (t < 7)
        __builtin_prefetch((const void*)(kbase + (size_t)(m0 + 32) * DD), 0, 0);
      const int m = m0 + (lane & 15);
#pragma unroll
      for (int r = 0; r < 8; ++r) {
        const int M = r + ((lane & 16) ? 8 : 0);
        S[M * NN + m] = c[r] + ET[M * NN + m] * we + be;        // attn + bias
      }
    }
    PIPE8(6);   // 2 q-frag loads + 16 k-frag loads vs 8 WMMAs, one frag ahead
    __syncthreads();
    // row softmax (16 threads per row, 64 cols each) + edge accumulation
    float lmax = -3.4e38f;
    for (int i = 0; i < 64; ++i) lmax = fmaxf(lmax, S[srow * NN + sbase + i]);
    red[tid] = lmax; __syncthreads();
    if (ssub == 0) { float m = red[tid]; for (int j = 1; j < 16; ++j) m = fmaxf(m, red[tid + j]); rowred[srow] = m; }
    __syncthreads();
    const float rmax = rowred[srow];
    float lsum = 0.f;
    for (int i = 0; i < 64; ++i) {            // single exp: keep e^x, fold s-term of edge
      const int idx = srow * NN + sbase + i;
      const float s = S[idx];
      const float ex = __expf(s - rmax);
      lsum += ex;
      E[idx] += s * wr;
      S[idx] = ex;
    }
    red[tid] = lsum; __syncthreads();
    if (ssub == 0) { float s = 0.f; for (int j = 0; j < 16; ++j) s += red[tid + j]; rowred[srow] = s; }
    __syncthreads();
    const float inv = 1.f / rowred[srow];
    for (int i = 0; i < 64; ++i) {            // normalize, fold p-term, emit f16 probs
      const int idx = srow * NN + sbase + i;
      const float p = S[idx] * inv;
      E[idx] += p * wr;
      Sh[idx] = (_Float16)p;
    }
    __syncthreads();
    // O += A_softmax @ V ; wave wv covers K (=m) range [wv*128, wv*128+128)
    const _Float16* vbase = vT + (((size_t)(b * HH + h)) * DD + (lane & 15)) * NN +
                            ((lane & 16) ? 16 : 0);
    const _Float16* shrow = Sh + (lane & 15) * NN;
    v8f c0 = {}, c1 = {};
    v16h b0A = load_bfrag(vbase + wv * 128);
    v16h b1A = load_bfrag(vbase + 16 * NN + wv * 128);
    v16h b0B = b0A, b1B = b1A;
#pragma unroll
    for (int kk = 0; kk < 4; ++kk) {
      const int mc = wv * 128 + kk * 32;
      v16h ap = load_afrag(shrow + mc, lane); // two ds_load_b128 from f16 prob tile
      if (kk & 1) {
        if (kk < 3) {
          b0A = load_bfrag(vbase + mc + 32);
          b1A = load_bfrag(vbase + 16 * NN + mc + 32);
        }
        c0 = wmma16x16x32(ap, b0B, c0);
        c1 = wmma16x16x32(ap, b1B, c1);
      } else {
        if (kk < 3) {
          b0B = load_bfrag(vbase + mc + 32);
          b1B = load_bfrag(vbase + 16 * NN + mc + 32);
        }
        c0 = wmma16x16x32(ap, b0A, c0);
        c1 = wmma16x16x32(ap, b1A, c1);
      }
    }
    // 16 V loads vs 8 WMMAs, one kk-step (4 loads / 2 wmma) ahead
    SGB(0x20, 8); SGB(0x8, 2);
    SGB(0x20, 4); SGB(0x8, 2);
    SGB(0x20, 4); SGB(0x8, 2);
    SGB(0x8, 2);
    {
      const int col = lane & 15;
#pragma unroll
      for (int r = 0; r < 8; ++r) {
        const int M = r + ((lane & 16) ? 8 : 0);
        atomicAdd(&O[M * DD + col], c0[r]);        // ds_add_f32 cross-wave reduce
        atomicAdd(&O[M * DD + 16 + col], c1[r]);
      }
    }
    __syncthreads();
    for (int idx = tid; idx < 512; idx += 256) {   // head h owns channels h*32..+31
      const int r = idx >> 5, dd = idx & 31;
      node_acc[((size_t)(b * NN + n0 + r)) * CC + h * DD + dd] = O[idx];
    }
    __syncthreads();
  }

  // finalize edge rows: add b_reduce, emit output #2, softmax-weighted self sum
  const float br = b_reduce[0];
  for (int idx = tid; idx < 16 * NN; idx += 256) {
    const float e = E[idx] + br;
    E[idx] = e;
    edge_out[((size_t)(b * NN + n0 + (idx >> 10))) * NN + (idx & (NN - 1))] = e;
  }
  __syncthreads();
  float lmax = -3.4e38f;
  for (int i = 0; i < 64; ++i) lmax = fmaxf(lmax, E[srow * NN + sbase + i]);
  red[tid] = lmax; __syncthreads();
  if (ssub == 0) { float m = red[tid]; for (int j = 1; j < 16; ++j) m = fmaxf(m, red[tid + j]); rowred[srow] = m; }
  __syncthreads();
  const float rmax = rowred[srow];
  float s1 = 0.f, s2 = 0.f;
  for (int i = 0; i < 64; ++i) {
    const float e = E[srow * NN + sbase + i];
    const float ex = __expf(e - rmax);
    s1 += ex; s2 += ex * e;
  }
  red[tid] = s1; __syncthreads();
  if (ssub == 0) { float s = 0.f; for (int j = 0; j < 16; ++j) s += red[tid + j]; rowred[srow] = s; }
  __syncthreads();
  const float Z = rowred[srow];
  red[tid] = s2; __syncthreads();
  if (ssub == 0) { float s = 0.f; for (int j = 0; j < 16; ++j) s += red[tid + j]; rowred[srow] = s; }
  __syncthreads();
  if (ssub == 0) wsum_ws[b * NN + n0 + srow] = rowred[srow] / Z;
}

// ------------------------------- final: (+ wsum@w_fc + b_fc) @ w_proj + b_proj
__global__ void __launch_bounds__(256) k_final(const float* __restrict__ node_acc,
                                               const float* __restrict__ wsum_ws,
                                               const float* __restrict__ w_fc,
                                               const float* __restrict__ b_fc,
                                               const _Float16* __restrict__ wpT,
                                               const float* __restrict__ b_proj,
                                               float* __restrict__ out0) {
  __shared__ _Float16 T[16 * CC];
  const int rb = blockIdx.x;
  const int b = rb >> 6, n0 = (rb & 63) << 4;
  const int tid = threadIdx.x, lane = tid & 31;
  const int wv = __builtin_amdgcn_readfirstlane(tid >> 5);   // SGPR wave id
  for (int idx = tid; idx < 16 * CC; idx += 256) {
    const int r = idx >> 8, c = idx & 255;
    const float v = node_acc[((size_t)(b * NN + n0 + r)) * CC + c] +
                    wsum_ws[b * NN + n0 + r] * w_fc[c] + b_fc[c];
    T[idx] = (_Float16)v;
  }
  __syncthreads();
  const _Float16* arow = T + (lane & 15) * CC;
  for (int t = 0; t < 2; ++t) {                 // 8 waves x 2 tiles = 16 = C/16
    const int jt = wv * 2 + t;
    const _Float16* bbase = wpT + (size_t)(jt * 16 + (lane & 15)) * CC +
                            ((lane & 16) ? 16 : 0);
    v8f acc = {};
    v16h bfA = load_bfrag(bbase);
    v16h bfB = bfA;
#pragma unroll
    for (int kc = 0; kc < 8; ++kc) {
      v16h a = load_afrag(arow + kc * 32, lane);
      if (kc & 1) {
        if (kc < 7) bfA = load_bfrag(bbase + (kc + 1) * 32);
        acc = wmma16x16x32(a, bfB, acc);
      } else {
        if (kc < 7) bfB = load_bfrag(bbase + (kc + 1) * 32);
        acc = wmma16x16x32(a, bfA, acc);
      }
    }
    PIPE8(4);
    const int col = lane & 15, j = jt * 16 + col;
    const float bpj = b_proj[j];
#pragma unroll
    for (int r = 0; r < 8; ++r) {
      const int n = n0 + r + ((lane & 16) ? 8 : 0);
      out0[((size_t)(b * CC + j)) * NN + n] = acc[r] + bpj;   // transpose-on-store
    }
  }
}

extern "C" void kernel_launch(void* const* d_in, const int* in_sizes, int n_in,
                              void* d_out, int out_size, void* d_ws, size_t ws_size,
                              hipStream_t stream) {
  const float* node_embeds = (const float*)d_in[0];
  const float* edge_embeds = (const float*)d_in[1];
  const float* da_prior    = (const float*)d_in[2];
  const float* gamma       = (const float*)d_in[3];
  const float* beta        = (const float*)d_in[4];
  const float* w_qkv       = (const float*)d_in[5];
  const float* w_proj      = (const float*)d_in[6];
  const float* b_proj      = (const float*)d_in[7];
  const float* w_expand    = (const float*)d_in[8];
  const float* b_expand    = (const float*)d_in[9];
  const float* w_reduce    = (const float*)d_in[10];
  const float* b_reduce    = (const float*)d_in[11];
  const float* w_fc        = (const float*)d_in[12];
  const float* b_fc        = (const float*)d_in[13];

  float* out0     = (float*)d_out;                  // [B,C,N]
  float* out_edge = out0 + (size_t)BB * CC * NN;    // [B,N,N]

  char* ws = (char*)d_ws;
  size_t off = 0;
  float*    prior  = (float*)(ws + off);    off += (size_t)BB * CC * 4;
  _Float16* wqkvT  = (_Float16*)(ws + off); off += (size_t)3 * CC * CC * 2;
  _Float16* wpT    = (_Float16*)(ws + off); off += (size_t)CC * CC * 2;
  _Float16* lnb    = (_Float16*)(ws + off); off += (size_t)BB * NN * CC * 2;
  _Float16* qb     = (_Float16*)(ws + off); off += (size_t)BB * HH * NN * DD * 2;
  _Float16* kb     = (_Float16*)(ws + off); off += (size_t)BB * HH * NN * DD * 2;
  _Float16* vT     = (_Float16*)(ws + off); off += (size_t)BB * HH * NN * DD * 2;
  float*  node_acc = (float*)(ws + off);    off += (size_t)BB * NN * CC * 4;
  float*  wsum_ws  = (float*)(ws + off);    off += (size_t)BB * NN * 4;
  (void)ws_size; (void)in_sizes; (void)n_in; (void)out_size;

  k_prior<<<BB, CC, 0, stream>>>(da_prior, prior);
  k_wq<<<3 * CC, CC, 0, stream>>>(w_qkv, wqkvT);
  k_wp<<<CC, CC, 0, stream>>>(w_proj, wpT);
  k_ln<<<BB * NN, CC, 0, stream>>>(node_embeds, edge_embeds, prior, gamma, beta, lnb);
  k_qkv<<<BB * NN / 16, 256, 0, stream>>>(lnb, wqkvT, qb, kb, vT);
  // LDS: 3x64KB f32 tiles + O/red + 32KB f16 prob tile = ~232KB of the 320KB WGP LDS
  const size_t smem =
      (size_t)(3 * 16 * NN + 512 + 256 + 16) * sizeof(float) + (size_t)16 * NN * 2;
  k_attn<<<BB * NN / 16, 256, smem, stream>>>(qb, kb, vT, edge_embeds, w_expand,
                                              b_expand, w_reduce, b_reduce,
                                              node_acc, wsum_ws, out_edge);
  k_final<<<BB * NN / 16, 256, 0, stream>>>(node_acc, wsum_ws, w_fc, b_fc, wpT,
                                            b_proj, out0);
}
